// WindowsSparseAttention_65893388255304
// MI455X (gfx1250) — matrix operand order repaired and checked by reading, not compiled, not run on hardware
//
#include <hip/hip_runtime.h>
#include <hip/hip_bf16.h>
#include <stdint.h>

// ---------------------------------------------------------------------------
// Windowed sparse attention for MI455X (gfx1250), wave32 + WMMA bf16.
//   q,k,v : (4, 8, 112, 112, 32) f32      indices : (4, 8, 256, 4) i32
//   out   : (4, 112, 112, 256) f32
// One block per (window, batch, head); 4 waves, each owning a 16-row Q tile.
// LDS strides padded so all B128 fragment loads tile the 64 LDS banks.
// ---------------------------------------------------------------------------

typedef __attribute__((ext_vector_type(16))) __bf16 v16bf;
typedef __attribute__((ext_vector_type(8)))  float  v8f;

namespace {
constexpr int kHeads  = 8;
constexpr int kB      = 4;
constexpr int kC      = 32;    // head channels
constexpr int kW2     = 49;    // tokens per 7x7 window
constexpr int kNW     = 256;   // windows (16x16)
constexpr int kTopK   = 4;
constexpr int kH      = 112;   // spatial extent
constexpr int kNKV    = 196;   // topk * 49 gathered K/V rows
constexpr int kNPadK  = 208;   // K rows padded to 13 tiles of 16
constexpr int kNPadP  = 224;   // P cols padded to 7 WMMA K-steps of 32
constexpr float kScale = 0.17677669529663687f;   // 32^-0.5 (folded into Q)

// Padded LDS row strides (elements). stride mod 64dw has gcd(.,64)=4 ->
// 16 distinct 4-bank-aligned starts -> conflict-free B128 fragment loads.
constexpr int kQStr = 40;      // bf16 per Q row   (20 dwords)
constexpr int kKStr = 40;      // bf16 per K row   (20 dwords)
constexpr int kVStr = 232;     // bf16 per V^T row (116 dwords)
constexpr int kPStr = 232;     // bf16 per P row   (116 dwords)
constexpr int kSStr = 228;     // f32  per S row   (228 dwords, 36 mod 64)

// LDS layout (dynamic shared memory, byte offsets — all 16B aligned)
constexpr int kOffS = 0;                          // 64*228*4 = 58368
constexpr int kOffQ = 58368;                      // 64*40*2  =  5120
constexpr int kOffK = 63488;                      // 208*40*2 = 16640
constexpr int kOffV = 80128;                      // 32*232*2 = 14848
constexpr int kOffP = 94976;                      // 64*232*2 = 29696
constexpr int kSmemBytes = 124672;
} // namespace

__device__ __forceinline__ unsigned short bf16c(float f) {
  union { float f; uint32_t u; } x; x.f = f;
  uint32_t r = x.u + 0x7FFFu + ((x.u >> 16) & 1u);   // round-to-nearest-even
  return (unsigned short)(r >> 16);
}

union Frag16 {               // 16 bf16 = 8 VGPRs (one WMMA A/B operand)
  v16bf v;
  uint4 u[2];
};

__device__ __forceinline__ v8f wmma_bf16(v16bf a, v16bf b, v8f c) {
  // D(f32 16x16) = A(bf16 16x32) * B(bf16 32x16) + C
  return __builtin_amdgcn_wmma_f32_16x16x32_bf16(
      /*neg_a=*/false, a, /*neg_b=*/false, b,
      /*c_mod=*/(short)0, c, /*reuse_a=*/false, /*reuse_b=*/false);
}

__global__ __launch_bounds__(128)
void win_sparse_attn_kernel(const float* __restrict__ q,
                            const float* __restrict__ k,
                            const float* __restrict__ v,
                            const int*   __restrict__ idx,
                            float*       __restrict__ out) {
  extern __shared__ char smem[];
  float*          Sb = (float*)(smem + kOffS);           // [64][kSStr] f32
  unsigned short* Qb = (unsigned short*)(smem + kOffQ);  // [64][kQStr] bf16
  unsigned short* Kb = (unsigned short*)(smem + kOffK);  // [208][kKStr] bf16
  unsigned short* Vt = (unsigned short*)(smem + kOffV);  // [32][kVStr] bf16 (V^T)
  unsigned short* Pb = (unsigned short*)(smem + kOffP);  // [64][kPStr] bf16

  const int tid  = threadIdx.x;
  const int wave = tid >> 5;
  const int lane = tid & 31;

  const int bid = blockIdx.x;
  const int h = bid & 7;
  const int b = (bid >> 3) & 3;
  const int w = bid >> 5;             // window id 0..255
  const int wy = w >> 4, wx = w & 15;

  const size_t headStride = (size_t)kH * kH * kC;        // 401408
  const float* qbase = q + ((size_t)b * kHeads + h) * headStride;
  const float* kbase = k + ((size_t)b * kHeads + h) * headStride;
  const float* vbase = v + ((size_t)b * kHeads + h) * headStride;
  const int*   ip    = idx + (((size_t)b * kHeads + h) * kNW + w) * kTopK;

  // ---------------- Phase 1: cooperative loads (f32 -> bf16 into LDS) ------
  // Q window: 49 rows x 32ch, pre-scaled by 1/sqrt(C); zero-pad rows 49..63.
  for (int i4 = tid; i4 < 64 * 8; i4 += 128) {           // float4 granules
    const int row = i4 >> 3, c4 = (i4 & 7) * 4;
    unsigned short* dst = &Qb[row * kQStr + c4];
    if (row < kW2) {
      const int y = wy * 7 + row / 7, x = wx * 7 + row % 7;
      const float4 val = *(const float4*)(qbase + ((size_t)y * kH + x) * kC + c4);
      dst[0] = bf16c(val.x * kScale); dst[1] = bf16c(val.y * kScale);
      dst[2] = bf16c(val.z * kScale); dst[3] = bf16c(val.w * kScale);
    } else {
      dst[0] = dst[1] = dst[2] = dst[3] = 0;
    }
  }
  // Gathered K (row-major) and V (transposed), zero-pad rows 196..207.
  for (int i4 = tid; i4 < kNPadK * 8; i4 += 128) {
    const int row = i4 >> 3, c4 = (i4 & 7) * 4;
    unsigned short* kd = &Kb[row * kKStr + c4];
    if (row < kNKV) {
      const int t = row / kW2, ri = row % kW2;
      const int wt = ip[t];
      const int ty = (wt >> 4) * 7 + ri / 7, tx = (wt & 15) * 7 + ri % 7;
      const size_t goff = ((size_t)ty * kH + tx) * kC + c4;
      const float4 kv = *(const float4*)(kbase + goff);
      kd[0] = bf16c(kv.x); kd[1] = bf16c(kv.y);
      kd[2] = bf16c(kv.z); kd[3] = bf16c(kv.w);
      const float4 vv = *(const float4*)(vbase + goff);
      Vt[(c4 + 0) * kVStr + row] = bf16c(vv.x);
      Vt[(c4 + 1) * kVStr + row] = bf16c(vv.y);
      Vt[(c4 + 2) * kVStr + row] = bf16c(vv.z);
      Vt[(c4 + 3) * kVStr + row] = bf16c(vv.w);
    } else {
      kd[0] = kd[1] = kd[2] = kd[3] = 0;
      Vt[(c4 + 0) * kVStr + row] = 0;
      Vt[(c4 + 1) * kVStr + row] = 0;
      Vt[(c4 + 2) * kVStr + row] = 0;
      Vt[(c4 + 3) * kVStr + row] = 0;
    }
  }
  // Zero V^T tail columns 208..223 (never filled above).
  for (int i = tid; i < kC * (kNPadP - kNPadK); i += 128) {
    const int c = i >> 4, r = kNPadK + (i & 15);
    Vt[c * kVStr + r] = 0;
  }
  __syncthreads();

  // ---------------- Phase 2: S = Q * K^T (13 WMMA per wave) ----------------
  const int m0   = wave * 16;              // this wave's Q-row tile
  const int am   = m0 + (lane & 15);       // A-frag row for this lane
  const int asel = (lane >> 4) * 8;        // A layout: lanes>=16 hold K+8
  Frag16 aq;                               // loop-invariant Q fragment
  aq.u[0] = *(const uint4*)&Qb[am * kQStr + asel];
  aq.u[1] = *(const uint4*)&Qb[am * kQStr + 16 + asel];

  const int koff = (lane >> 4) * 8;        // B layout: lanes>=16 hold K+8
  auto loadK = [&](int nt, Frag16& f) {
    const int n = nt * 16 + (lane & 15);   // B column = gathered K row
    f.u[0] = *(const uint4*)&Kb[n * kKStr + koff];
    f.u[1] = *(const uint4*)&Kb[n * kKStr + koff + 16];
  };
  Frag16 bkA, bkB;
  loadK(0, bkA);
  const int sr = m0 + 8 * (lane >> 4);     // C layout row base for this lane
#pragma unroll
  for (int nt = 0; nt < kNPadK / 16; ++nt) {
    Frag16& cur = (nt & 1) ? bkB : bkA;
    Frag16& nxt = (nt & 1) ? bkA : bkB;
    if (nt + 1 < kNPadK / 16) loadK(nt + 1, nxt);   // double-buffer prefetch
    v8f acc = {};
    acc = wmma_bf16(aq.v, cur.v, acc);
    const int sc = nt * 16 + (lane & 15);
#pragma unroll
    for (int r = 0; r < 8; ++r) Sb[(sr + r) * kSStr + sc] = acc[r];
  }
  __syncthreads();

  // ---------------- Phase 3: softmax (lane pairs split each row) -----------
  {
    const int row  = m0 + (lane & 15);
    const int half = lane >> 4;
    const int g0 = half ? 24 : 0;          // float4 granules: 196 = 49*4
    const int g1 = half ? 49 : 24;
    float* srow = &Sb[row * kSStr];
    float mx = -3.0e38f;
    for (int g = g0; g < g1; ++g) {
      const float4 s4 = *(const float4*)&srow[g * 4];
      mx = fmaxf(mx, fmaxf(fmaxf(s4.x, s4.y), fmaxf(s4.z, s4.w)));
    }
    mx = fmaxf(mx, __shfl_xor(mx, 16, 32));
    float sum = 0.f;
    for (int g = g0; g < g1; ++g) {
      float4 s4 = *(const float4*)&srow[g * 4];
      s4.x = __expf(s4.x - mx); s4.y = __expf(s4.y - mx);
      s4.z = __expf(s4.z - mx); s4.w = __expf(s4.w - mx);
      *(float4*)&srow[g * 4] = s4;
      sum += (s4.x + s4.y) + (s4.z + s4.w);
    }
    sum += __shfl_xor(sum, 16, 32);
    const float inv = 1.0f / sum;
    unsigned short* prow = &Pb[row * kPStr];
    for (int g = g0; g < g1; ++g) {
      const float4 s4 = *(const float4*)&srow[g * 4];
      uint2 pk;
      pk.x = (uint32_t)bf16c(s4.x * inv) | ((uint32_t)bf16c(s4.y * inv) << 16);
      pk.y = (uint32_t)bf16c(s4.z * inv) | ((uint32_t)bf16c(s4.w * inv) << 16);
      *(uint2*)&prow[g * 4] = pk;
    }
    if (half) {                             // zero padded columns 196..223
      for (int c = kNKV; c < kNPadP; ++c) prow[c] = 0;
    }
  }
  __syncthreads();

  // ---------------- Phase 4: O = P * V (7 K-steps x 2 N-tiles) -------------
  auto loadPV = [&](int kk, Frag16& ap, Frag16& b0, Frag16& b1) {
    ap.u[0] = *(const uint4*)&Pb[am * kPStr + kk + asel];
    ap.u[1] = *(const uint4*)&Pb[am * kPStr + kk + 16 + asel];
    const int kb = kk + koff;
    b0.u[0] = *(const uint4*)&Vt[(lane & 15) * kVStr + kb];
    b0.u[1] = *(const uint4*)&Vt[(lane & 15) * kVStr + kb + 16];
    b1.u[0] = *(const uint4*)&Vt[(16 + (lane & 15)) * kVStr + kb];
    b1.u[1] = *(const uint4*)&Vt[(16 + (lane & 15)) * kVStr + kb + 16];
  };
  v8f o0 = {}, o1 = {};
  Frag16 apA, apB, bv0A, bv0B, bv1A, bv1B;
  loadPV(0, apA, bv0A, bv1A);
#pragma unroll
  for (int t = 0; t < kNPadP / 32; ++t) {
    Frag16& ap  = (t & 1) ? apB  : apA;
    Frag16& b0  = (t & 1) ? bv0B : bv0A;
    Frag16& b1  = (t & 1) ? bv1B : bv1A;
    Frag16& apn = (t & 1) ? apA  : apB;
    Frag16& b0n = (t & 1) ? bv0A : bv0B;
    Frag16& b1n = (t & 1) ? bv1A : bv1B;
    if (t + 1 < kNPadP / 32) loadPV((t + 1) * 32, apn, b0n, b1n);
    o0 = wmma_bf16(ap.v, b0.v, o0);
    o1 = wmma_bf16(ap.v, b1.v, o1);
  }

  // ---------------- Phase 5: store output ----------------------------------
  // out[b][y][x][h*32 + c], 16 consecutive channels per lane half.
  float* obase = out + (size_t)b * kH * kH * (kHeads * kC) + h * kC;
  const int orow0 = m0 + 8 * (lane >> 4);
  const int oc    = lane & 15;
#pragma unroll
  for (int r = 0; r < 8; ++r) {
    const int row = orow0 + r;
    if (row < kW2) {
      const int y = wy * 7 + row / 7, x = wx * 7 + row % 7;
      float* p = obase + ((size_t)y * kH + x) * (kHeads * kC);
      p[oc]      = o0[r];
      p[16 + oc] = o1[r];
    }
  }
}

extern "C" void kernel_launch(void* const* d_in, const int* in_sizes, int n_in,
                              void* d_out, int out_size, void* d_ws, size_t ws_size,
                              hipStream_t stream) {
  (void)in_sizes; (void)n_in; (void)out_size; (void)d_ws; (void)ws_size;
  const float* q   = (const float*)d_in[0];
  const float* k   = (const float*)d_in[1];
  const float* v   = (const float*)d_in[2];
  const int*   idx = (const int*)d_in[3];
  float*       out = (float*)d_out;

  // Allow >64KB dynamic LDS (CDNA5 WGP has 320KB).
  hipFuncSetAttribute((const void*)win_sparse_attn_kernel,
                      hipFuncAttributeMaxDynamicSharedMemorySize, kSmemBytes);

  const dim3 grid(kNW * kB * kHeads);   // 8192 blocks: one per (w, b, h)
  win_sparse_attn_kernel<<<grid, 128, kSmemBytes, stream>>>(q, k, v, idx, out);
}
